// Anchors_47132971106494
// MI455X (gfx1250) — matrix operand order repaired and checked by reading, not compiled, not run on hardware
//
#include <hip/hip_runtime.h>
#include <hip/hip_bf16.h>
#include <math.h>

// Problem dims (from reference): B=4096, C=100, A=64, E=768 ; N = C*A = 6400
#define BB 4096
#define NN 6400
#define EE 768

typedef __bf16 bf16_t;
typedef __attribute__((ext_vector_type(8)))  __bf16 v8bf;
typedef __attribute__((ext_vector_type(16))) __bf16 v16bf;
typedef __attribute__((ext_vector_type(8)))  float  v8f;

// ---------------------------------------------------------------------------
// Phase 1: fp32 -> bf16 hi/lo split + row squared-norms
// One 256-thread block per row of `ncols` (=768) elements.
// ---------------------------------------------------------------------------
__global__ __launch_bounds__(256)
void split_rows(const float* __restrict__ src,
                bf16_t* __restrict__ hi, bf16_t* __restrict__ lo,
                float* __restrict__ nrm, int ncols) {
  __shared__ float red[256];
  const int row = blockIdx.x;
  const float*  s = src + (size_t)row * ncols;
  bf16_t* h = hi + (size_t)row * ncols;
  bf16_t* l = lo + (size_t)row * ncols;
  float acc = 0.0f;
  for (int e = threadIdx.x; e < ncols; e += 256) {
    float v  = s[e];
    __bf16 vh = (__bf16)v;            // RNE truncation to bf16 (8-bit mantissa)
    float  rm = v - (float)vh;        // exact residual
    __bf16 vl = (__bf16)rm;           // next 8 mantissa bits
    h[e] = vh;
    l[e] = vl;
    acc += v * v;                     // norms in full fp32
  }
  red[threadIdx.x] = acc;
  __syncthreads();
  for (int off = 128; off > 0; off >>= 1) {
    if (threadIdx.x < off) red[threadIdx.x] += red[threadIdx.x + off];
    __syncthreads();
  }
  if (threadIdx.x == 0) nrm[row] = red[0];
}

// ---------------------------------------------------------------------------
// Fragment loaders (wave32 WMMA layouts, 16-bit data)
// A 16x32: lane holds M=lane&15; K-octets at kbase=half*8 and kbase+16
// B 32x16: lane holds N=lane&15; K-octets at kbase=half*8 and kbase+16
// Both are two 16-byte loads from K-contiguous row-major storage.
// ---------------------------------------------------------------------------
__device__ __forceinline__ v16bf ldA(const bf16_t* p) {
  v8bf c0 = *(const v8bf*)(p);
  v8bf c1 = *(const v8bf*)(p + 16);
  return __builtin_shufflevector(c0, c1, 0,1,2,3,4,5,6,7,8,9,10,11,12,13,14,15);
}
__device__ __forceinline__ v16bf ldB(const bf16_t* p) {
  v8bf c0 = *(const v8bf*)(p);
  v8bf c1 = *(const v8bf*)(p + 16);
  return __builtin_shufflevector(c0, c1, 0,1,2,3,4,5,6,7,8,9,10,11,12,13,14,15);
}

#define WMMA_ACC(Cacc, Af, Bf)                                                \
  Cacc = __builtin_amdgcn_wmma_f32_16x16x32_bf16(false, (Af), false, (Bf),    \
                                                 (short)0, (Cacc), false, false)

// ---------------------------------------------------------------------------
// Phase 2: distance GEMM.
// Block = 256 threads = 8 waves, arranged 4 (M) x 2 (N); each wave computes a
// 32x32 output tile via 2x2 of 16x16 WMMA tiles, bf16x3 split-precision.
// Grid: (NN/64, BB/128).
// ---------------------------------------------------------------------------
__global__ __launch_bounds__(256)
void dist_wmma(const bf16_t* __restrict__ xh, const bf16_t* __restrict__ xl,
               const bf16_t* __restrict__ ah, const bf16_t* __restrict__ al,
               const float* __restrict__ x2, const float* __restrict__ a2,
               float* __restrict__ out) {
  const int lane = threadIdx.x & 31;
  const int wave = threadIdx.x >> 5;
  const int wm = wave & 3;            // wave position in M
  const int wn = wave >> 2;           // wave position in N
  const int m0 = blockIdx.y * 128 + wm * 32;
  const int n0 = blockIdx.x * 64  + wn * 32;
  const int li   = lane & 15;
  const int half = lane >> 4;

  // Per-lane base pointers into K-contiguous rows.
  const bf16_t* pxh0 = xh + (size_t)(m0      + li) * EE + half * 8;
  const bf16_t* pxh1 = xh + (size_t)(m0 + 16 + li) * EE + half * 8;
  const bf16_t* pxl0 = xl + (size_t)(m0      + li) * EE + half * 8;
  const bf16_t* pxl1 = xl + (size_t)(m0 + 16 + li) * EE + half * 8;
  const bf16_t* pah0 = ah + (size_t)(n0      + li) * EE + half * 8;
  const bf16_t* pah1 = ah + (size_t)(n0 + 16 + li) * EE + half * 8;
  const bf16_t* pal0 = al + (size_t)(n0      + li) * EE + half * 8;
  const bf16_t* pal1 = al + (size_t)(n0 + 16 + li) * EE + half * 8;

  v8f c00 = {}, c01 = {}, c10 = {}, c11 = {};

  for (int k = 0; k < EE; k += 32) {
    v16bf a0H = ldA(pxh0 + k);
    v16bf a1H = ldA(pxh1 + k);
    v16bf a0L = ldA(pxl0 + k);
    v16bf a1L = ldA(pxl1 + k);
    v16bf b0H = ldB(pah0 + k);
    v16bf b1H = ldB(pah1 + k);
    v16bf b0L = ldB(pal0 + k);
    v16bf b1L = ldB(pal1 + k);

    // bf16x3: hi*hi + hi*lo + lo*hi, fp32 accumulate
    WMMA_ACC(c00, a0H, b0H); WMMA_ACC(c00, a0H, b0L); WMMA_ACC(c00, a0L, b0H);
    WMMA_ACC(c01, a0H, b1H); WMMA_ACC(c01, a0H, b1L); WMMA_ACC(c01, a0L, b1H);
    WMMA_ACC(c10, a1H, b0H); WMMA_ACC(c10, a1H, b0L); WMMA_ACC(c10, a1L, b0H);
    WMMA_ACC(c11, a1H, b1H); WMMA_ACC(c11, a1H, b1L); WMMA_ACC(c11, a1L, b1H);
  }

  // Epilogue: d = sqrt(max(x2[m] + a2[n] - 2*dot, 0))
  // C/D layout: lane holds N=li; VGPR v holds row m16 + half*8 + v.
  const int rb0 = m0 + half * 8;
  const int rb1 = m0 + 16 + half * 8;
  const int col0 = n0 + li;
  const int col1 = n0 + 16 + li;
  const float a2c0 = a2[col0];
  const float a2c1 = a2[col1];
  const float* x2p0 = x2 + rb0;
  const float* x2p1 = x2 + rb1;

#pragma unroll
  for (int v = 0; v < 8; ++v) {
    float xn0 = x2p0[v];
    float xn1 = x2p1[v];
    float d00 = xn0 + a2c0 - 2.0f * c00[v];
    float d01 = xn0 + a2c1 - 2.0f * c01[v];
    float d10 = xn1 + a2c0 - 2.0f * c10[v];
    float d11 = xn1 + a2c1 - 2.0f * c11[v];
    out[(size_t)(rb0 + v) * NN + col0] = sqrtf(fmaxf(d00, 0.0f));
    out[(size_t)(rb0 + v) * NN + col1] = sqrtf(fmaxf(d01, 0.0f));
    out[(size_t)(rb1 + v) * NN + col0] = sqrtf(fmaxf(d10, 0.0f));
    out[(size_t)(rb1 + v) * NN + col1] = sqrtf(fmaxf(d11, 0.0f));
  }
}

// ---------------------------------------------------------------------------
// Launch. Workspace layout (bytes), all 16B-aligned:
//   xh:        0 .. 6291456      (4096*768 bf16)
//   xl:  6291456 .. 12582912
//   ah: 12582912 .. 22413312     (6400*768 bf16)
//   al: 22413312 .. 32243712
//   x2: 32243712 .. 32260096     (4096 f32)
//   a2: 32260096 .. 32285696     (6400 f32)
// Total ~32.3 MB of d_ws.
// ---------------------------------------------------------------------------
extern "C" void kernel_launch(void* const* d_in, const int* in_sizes, int n_in,
                              void* d_out, int out_size, void* d_ws, size_t ws_size,
                              hipStream_t stream) {
  const float* x       = (const float*)d_in[0];   // [4096, 768]
  const float* anchors = (const float*)d_in[1];   // [100, 64, 768] -> [6400, 768]
  float* out = (float*)d_out;                     // [4096, 100, 64]

  char* ws = (char*)d_ws;
  bf16_t* xh = (bf16_t*)(ws + 0);
  bf16_t* xl = (bf16_t*)(ws + 6291456);
  bf16_t* ah = (bf16_t*)(ws + 12582912);
  bf16_t* al = (bf16_t*)(ws + 22413312);
  float*  x2 = (float*)(ws + 32243712);
  float*  a2 = (float*)(ws + 32260096);

  split_rows<<<BB, 256, 0, stream>>>(x, xh, xl, x2, EE);
  split_rows<<<NN, 256, 0, stream>>>(anchors, ah, al, a2, EE);

  dim3 grid(NN / 64, BB / 128);  // (100, 32)
  dist_wmma<<<grid, 256, 0, stream>>>(xh, xl, ah, al, x2, a2, out);
}
